// MultiDomainLeNet_61667140436470
// MI455X (gfx1250) — matrix operand order repaired and checked by reading, not compiled, not run on hardware
//
#include <hip/hip_runtime.h>
#include <hip/hip_bf16.h>

typedef _Float16 f16;
typedef __attribute__((ext_vector_type(8)))  _Float16 half8;
typedef __attribute__((ext_vector_type(16))) _Float16 v16h;
typedef __attribute__((ext_vector_type(8)))  float    v8f;

#define BATCH 16384
#define NDOM  3
#define EPSBN 1e-5f

// ---------------------------------------------------------------------------
// Weight prep: f32 [Dn][K][N] -> f16 [Dn][Np][Kp] (transposed, zero padded)
// ---------------------------------------------------------------------------
__global__ void prep_wt(const float* __restrict__ src, f16* __restrict__ dst,
                        int Dn, int K, int N, int Kp, int Np) {
    int total = Dn * Np * Kp;
    for (int i = blockIdx.x * blockDim.x + threadIdx.x; i < total;
         i += gridDim.x * blockDim.x) {
        int k  = i % Kp;
        int n  = (i / Kp) % Np;
        int dz = i / (Kp * Np);
        float v = (k < K && n < N) ? src[((size_t)dz * K + k) * N + n] : 0.0f;
        dst[i] = (f16)v;
    }
}

// ---------------------------------------------------------------------------
// Stage 1: conv1 (1->6, 3x3 SAME, 28x28) per-sample sum / sumsq partials
// ---------------------------------------------------------------------------
__global__ void conv1_stats(const float* __restrict__ x,
                            const float* __restrict__ w,
                            const float* __restrict__ bias,
                            float* __restrict__ psum, float* __restrict__ psq) {
    int b = blockIdx.x, tid = threadIdx.x;
    __shared__ float xs[784];
    __shared__ float cw[54], cb[6];
    __shared__ float rA[256], rB[256];
    for (int i = tid; i < 784; i += 256) xs[i] = x[(size_t)b * 784 + i];
    if (tid < 54) cw[tid] = w[tid];
    if (tid < 6)  cb[tid] = bias[tid];
    __syncthreads();
    for (int c = 0; c < 6; ++c) {
        float s = 0.f, q = 0.f;
        for (int p = tid; p < 784; p += 256) {
            int y = p / 28, xc = p % 28;
            float acc = cb[c];
            #pragma unroll
            for (int t = 0; t < 9; ++t) {
                int yy = y + t / 3 - 1, xx = xc + t % 3 - 1;
                if ((unsigned)yy < 28u && (unsigned)xx < 28u)
                    acc += cw[c * 9 + t] * xs[yy * 28 + xx];
            }
            s += acc; q += acc * acc;
        }
        rA[tid] = s; rB[tid] = q; __syncthreads();
        for (int off = 128; off > 0; off >>= 1) {
            if (tid < off) { rA[tid] += rA[tid + off]; rB[tid] += rB[tid + off]; }
            __syncthreads();
        }
        if (tid == 0) { psum[b * 6 + c] = rA[0]; psq[b * 6 + c] = rB[0]; }
        __syncthreads();
    }
}

// ---------------------------------------------------------------------------
// BN finalize: masked deterministic reduction over batch -> scale/shift
// ---------------------------------------------------------------------------
__global__ void bn_finalize(const float* __restrict__ psum,
                            const float* __restrict__ psq,
                            const int* __restrict__ dom,
                            const float* __restrict__ gamma,
                            const float* __restrict__ beta,
                            float* __restrict__ scale, float* __restrict__ shift,
                            int C, float hw) {
    int d = blockIdx.x, c = blockIdx.y, tid = threadIdx.x;
    __shared__ float rA[256], rB[256], rC[256];
    float s = 0.f, q = 0.f, cnt = 0.f;
    for (int i = tid; i < BATCH; i += 256) {
        if (dom[i] == d) { s += psum[(size_t)i * C + c]; q += psq[(size_t)i * C + c]; cnt += 1.f; }
    }
    rA[tid] = s; rB[tid] = q; rC[tid] = cnt; __syncthreads();
    for (int off = 128; off > 0; off >>= 1) {
        if (tid < off) { rA[tid] += rA[tid+off]; rB[tid] += rB[tid+off]; rC[tid] += rC[tid+off]; }
        __syncthreads();
    }
    if (tid == 0) {
        float n    = fmaxf(rC[0] * hw, 1.0f);
        float mean = rA[0] / n;
        float var  = fmaxf(rB[0] / n - mean * mean, 0.0f);
        float sc   = gamma[d * C + c] * rsqrtf(var + EPSBN);
        scale[d * C + c] = sc;
        shift[d * C + c] = beta[d * C + c] - mean * sc;
    }
}

// ---------------------------------------------------------------------------
// Stage 2: conv1 -> BN1 -> relu -> pool (store h1p) -> conv2 stats partials
// ---------------------------------------------------------------------------
__global__ void conv1_apply_conv2_stats(const float* __restrict__ x,
                                        const int* __restrict__ dom,
                                        const float* __restrict__ w1,
                                        const float* __restrict__ b1,
                                        const float* __restrict__ sc1g,
                                        const float* __restrict__ sh1g,
                                        const float* __restrict__ w2,
                                        const float* __restrict__ b2,
                                        float* __restrict__ h1p_g,
                                        float* __restrict__ psum2,
                                        float* __restrict__ psq2) {
    int b = blockIdx.x, tid = threadIdx.x;
    int d = dom[b];
    __shared__ float xs[784];
    __shared__ float hp[1176];                 // 6 x 14 x 14 pooled
    __shared__ float cw1[54], cb1[6], s1[6], h1[6];
    __shared__ float cw2[864], cb2[16];
    __shared__ float rA[256], rB[256];
    for (int i = tid; i < 784; i += 256) xs[i] = x[(size_t)b * 784 + i];
    if (tid < 54) cw1[tid] = w1[tid];
    if (tid < 6) { cb1[tid] = b1[tid]; s1[tid] = sc1g[d * 6 + tid]; h1[tid] = sh1g[d * 6 + tid]; }
    for (int i = tid; i < 864; i += 256) cw2[i] = w2[i];
    if (tid < 16) cb2[tid] = b2[tid];
    __syncthreads();
    // conv1 + bn + relu + 2x2 maxpool, evaluated directly at pooled pixels
    for (int o = tid; o < 1176; o += 256) {
        int c = o / 196, r = o % 196, py = r / 14, px = r % 14;
        float mx = 0.0f;                       // relu floor
        #pragma unroll
        for (int s = 0; s < 4; ++s) {
            int y = py * 2 + (s >> 1), xc = px * 2 + (s & 1);
            float acc = cb1[c];
            #pragma unroll
            for (int t = 0; t < 9; ++t) {
                int yy = y + t / 3 - 1, xx = xc + t % 3 - 1;
                if ((unsigned)yy < 28u && (unsigned)xx < 28u)
                    acc += cw1[c * 9 + t] * xs[yy * 28 + xx];
            }
            mx = fmaxf(mx, fmaxf(acc * s1[c] + h1[c], 0.0f));
        }
        hp[o] = mx;
        h1p_g[(size_t)b * 1176 + o] = mx;
    }
    __syncthreads();
    // conv2 (6->16, 3x3 SAME, 14x14) stats partials
    for (int c = 0; c < 16; ++c) {
        float s = 0.f, q = 0.f;
        if (tid < 196) {
            int y = tid / 14, xc = tid % 14;
            float acc = cb2[c];
            for (int ci = 0; ci < 6; ++ci)
                #pragma unroll
                for (int t = 0; t < 9; ++t) {
                    int yy = y + t / 3 - 1, xx = xc + t % 3 - 1;
                    if ((unsigned)yy < 14u && (unsigned)xx < 14u)
                        acc += cw2[(c * 6 + ci) * 9 + t] * hp[ci * 196 + yy * 14 + xx];
                }
            s = acc; q = acc * acc;
        }
        rA[tid] = s; rB[tid] = q; __syncthreads();
        for (int off = 128; off > 0; off >>= 1) {
            if (tid < off) { rA[tid] += rA[tid + off]; rB[tid] += rB[tid + off]; }
            __syncthreads();
        }
        if (tid == 0) { psum2[b * 16 + c] = rA[0]; psq2[b * 16 + c] = rB[0]; }
        __syncthreads();
    }
}

// ---------------------------------------------------------------------------
// Stage 3: conv2 -> BN2 -> relu -> pool -> feats f16 (row stride 800, padded)
// ---------------------------------------------------------------------------
__global__ void conv2_apply(const float* __restrict__ h1p_g,
                            const int* __restrict__ dom,
                            const float* __restrict__ w2,
                            const float* __restrict__ b2,
                            const float* __restrict__ sc2g,
                            const float* __restrict__ sh2g,
                            f16* __restrict__ feats) {
    int b = blockIdx.x, tid = threadIdx.x;
    int d = dom[b];
    __shared__ float hp[1176];
    __shared__ float cw2[864], cb2[16], s2[16], h2[16];
    for (int i = tid; i < 1176; i += 256) hp[i] = h1p_g[(size_t)b * 1176 + i];
    for (int i = tid; i < 864; i += 256) cw2[i] = w2[i];
    if (tid < 16) { cb2[tid] = b2[tid]; s2[tid] = sc2g[d * 16 + tid]; h2[tid] = sh2g[d * 16 + tid]; }
    __syncthreads();
    for (int o = tid; o < 784; o += 256) {
        int c = o / 49, r = o % 49, py = r / 7, px = r % 7;
        float mx = 0.0f;
        #pragma unroll
        for (int s = 0; s < 4; ++s) {
            int y = py * 2 + (s >> 1), xc = px * 2 + (s & 1);
            float acc = cb2[c];
            for (int ci = 0; ci < 6; ++ci)
                #pragma unroll
                for (int t = 0; t < 9; ++t) {
                    int yy = y + t / 3 - 1, xx = xc + t % 3 - 1;
                    if ((unsigned)yy < 14u && (unsigned)xx < 14u)
                        acc += cw2[(c * 6 + ci) * 9 + t] * hp[ci * 196 + yy * 14 + xx];
                }
            mx = fmaxf(mx, fmaxf(acc * s2[c] + h2[c], 0.0f));
        }
        feats[(size_t)b * 800 + o] = (f16)mx;
    }
    for (int o = 784 + tid; o < 800; o += 256) feats[(size_t)b * 800 + o] = (f16)0.0f;
}

// ---------------------------------------------------------------------------
// WMMA 16x16 tile: A [16 x 32*kTiles] row-major, Bt [16 x 32*kTiles] (N-major,
// i.e. B transposed). Fragment layouts per CDNA5 ISA 16-bit A / B tables.
// ---------------------------------------------------------------------------
__device__ __forceinline__ v8f wmma_tile_f16(const f16* __restrict__ A, int lda,
                                             const f16* __restrict__ Bt, int ldb,
                                             int kTiles) {
    int lane = threadIdx.x & 31;
    int hi   = lane >> 4;       // lane group (0: lanes 0-15, 1: lanes 16-31)
    int l16  = lane & 15;
    v8f c = {0.f, 0.f, 0.f, 0.f, 0.f, 0.f, 0.f, 0.f};
    const f16* arow = A  + (size_t)l16 * lda + hi * 8;   // A: M=l16; K chunks {k0..k0+7, k0+16..k0+23}
    const f16* brow = Bt + (size_t)l16 * ldb + hi * 8;   // B: N=l16; K chunks {k0..k0+7, k0+16..k0+23}
    for (int kk = 0; kk < kTiles; ++kk) {
        half8 a0 = *(const half8*)(arow + kk * 32);
        half8 a1 = *(const half8*)(arow + kk * 32 + 16);
        half8 b0 = *(const half8*)(brow + kk * 32);
        half8 b1 = *(const half8*)(brow + kk * 32 + 16);
        v16h av = __builtin_shufflevector(a0, a1, 0,1,2,3,4,5,6,7,8,9,10,11,12,13,14,15);
        v16h bv = __builtin_shufflevector(b0, b1, 0,1,2,3,4,5,6,7,8,9,10,11,12,13,14,15);
        c = __builtin_amdgcn_wmma_f32_16x16x32_f16(false, av, false, bv,
                                                   (short)0, c, false, false);
    }
    return c;
}

// GEMM + bias + relu -> f16 (pad columns written as exact zeros)
__global__ void gemm_relu_f16(const f16* __restrict__ A, long long sA, int lda,
                              const f16* __restrict__ Bt, long long sB, int ldb,
                              const float* __restrict__ bias, long long sBias,
                              int nValid, f16* __restrict__ out, long long sOut,
                              int ldo, int kTiles) {
    int z = blockIdx.z;
    const f16* At = A  + (size_t)sA * z + (size_t)blockIdx.x * 16 * lda;
    const f16* Bz = Bt + (size_t)sB * z + (size_t)blockIdx.y * 16 * ldb;
    const float* bz = bias + (size_t)sBias * z;
    f16* oz = out + (size_t)sOut * z;
    v8f c = wmma_tile_f16(At, lda, Bz, ldb, kTiles);
    int lane = threadIdx.x & 31, hi = lane >> 4, l16 = lane & 15;
    int colg = blockIdx.y * 16 + l16;
    #pragma unroll
    for (int r = 0; r < 8; ++r) {
        int row = blockIdx.x * 16 + hi * 8 + r;
        float v = (colg < nValid) ? fmaxf(c[r] + bz[colg], 0.0f) : 0.0f;
        oz[(size_t)row * ldo + colg] = (f16)v;
    }
}

// Final classifier GEMM fused with the per-sample domain gather
__global__ void gemm_logits_gather(const f16* __restrict__ A, long long sA, int lda,
                                   const f16* __restrict__ Bt, long long sB, int ldb,
                                   const float* __restrict__ bias, long long sBias,
                                   const int* __restrict__ dom,
                                   float* __restrict__ out, int nValid, int kTiles) {
    int z = blockIdx.z;
    const f16* At = A  + (size_t)sA * z + (size_t)blockIdx.x * 16 * lda;
    const f16* Bz = Bt + (size_t)sB * z;
    const float* bz = bias + (size_t)sBias * z;
    v8f c = wmma_tile_f16(At, lda, Bz, ldb, kTiles);
    int lane = threadIdx.x & 31, hi = lane >> 4, l16 = lane & 15;
    #pragma unroll
    for (int r = 0; r < 8; ++r) {
        int row = blockIdx.x * 16 + hi * 8 + r;
        if (l16 < nValid && dom[row] == z)
            out[(size_t)row * nValid + l16] = c[r] + bz[l16];
    }
}

// Final discriminator GEMM -> f32 (no relu)
__global__ void gemm_out_f32(const f16* __restrict__ A, int lda,
                             const f16* __restrict__ Bt, int ldb,
                             const float* __restrict__ bias,
                             float* __restrict__ out, int nValid, int kTiles) {
    const f16* At = A + (size_t)blockIdx.x * 16 * lda;
    v8f c = wmma_tile_f16(At, lda, Bt, ldb, kTiles);
    int lane = threadIdx.x & 31, hi = lane >> 4, l16 = lane & 15;
    #pragma unroll
    for (int r = 0; r < 8; ++r) {
        int row = blockIdx.x * 16 + hi * 8 + r;
        if (l16 < nValid)
            out[(size_t)row * nValid + l16] = c[r] + bias[l16];
    }
}

// ---------------------------------------------------------------------------
extern "C" void kernel_launch(void* const* d_in, const int* in_sizes, int n_in,
                              void* d_out, int out_size, void* d_ws, size_t ws_size,
                              hipStream_t stream) {
    (void)in_sizes; (void)n_in; (void)out_size; (void)ws_size;
    const float* x        = (const float*)d_in[0];
    const int*   dom      = (const int*)  d_in[1];
    const float* conv1_w  = (const float*)d_in[2];
    const float* conv1_b  = (const float*)d_in[3];
    const float* bn1_g    = (const float*)d_in[4];
    const float* bn1_b    = (const float*)d_in[5];
    const float* conv2_w  = (const float*)d_in[6];
    const float* conv2_b  = (const float*)d_in[7];
    const float* bn2_g    = (const float*)d_in[8];
    const float* bn2_b    = (const float*)d_in[9];
    const float* cls_w1   = (const float*)d_in[10];
    const float* cls_b1   = (const float*)d_in[11];
    const float* cls_w2   = (const float*)d_in[12];
    const float* cls_b2   = (const float*)d_in[13];
    const float* cls_w3   = (const float*)d_in[14];
    const float* cls_b3   = (const float*)d_in[15];
    const float* disc_w1  = (const float*)d_in[16];
    const float* disc_b1  = (const float*)d_in[17];
    const float* disc_w2  = (const float*)d_in[18];
    const float* disc_b2  = (const float*)d_in[19];
    float* out = (float*)d_out;

    char*  ws  = (char*)d_ws;
    size_t off = 0;
    auto alloc = [&](size_t bytes) -> char* {
        char* p = ws + off;
        off = (off + bytes + 255) & ~(size_t)255;
        return p;
    };
    float* h1p   = (float*)alloc((size_t)BATCH * 1176 * 4);      // 77 MB
    f16*   feats = (f16*)  alloc((size_t)BATCH * 800 * 2);       // 26 MB
    f16*   a1    = (f16*)  alloc((size_t)NDOM * BATCH * 128 * 2);
    f16*   a2    = (f16*)  alloc((size_t)NDOM * BATCH * 96 * 2);
    f16*   hd    = (f16*)  alloc((size_t)BATCH * 128 * 2);
    f16*   w1t   = (f16*)  alloc((size_t)NDOM * 128 * 800 * 2);
    f16*   w2t   = (f16*)  alloc((size_t)NDOM * 96 * 128 * 2);
    f16*   w3t   = (f16*)  alloc((size_t)NDOM * 16 * 96 * 2);
    f16*   dw1t  = (f16*)  alloc((size_t)128 * 800 * 2);
    f16*   dw2t  = (f16*)  alloc((size_t)16 * 128 * 2);
    float* ps1s  = (float*)alloc((size_t)BATCH * 6 * 4);
    float* ps1q  = (float*)alloc((size_t)BATCH * 6 * 4);
    float* ps2s  = (float*)alloc((size_t)BATCH * 16 * 4);
    float* ps2q  = (float*)alloc((size_t)BATCH * 16 * 4);
    float* sc1   = (float*)alloc(NDOM * 6 * 4);
    float* sh1   = (float*)alloc(NDOM * 6 * 4);
    float* sc2   = (float*)alloc(NDOM * 16 * 4);
    float* sh2   = (float*)alloc(NDOM * 16 * 4);

    // Weight prep (transpose + pad + f16)
    prep_wt<<<dim3(1200), dim3(256), 0, stream>>>(cls_w1, w1t, NDOM, 784, 120, 800, 128);
    prep_wt<<<dim3(144),  dim3(256), 0, stream>>>(cls_w2, w2t, NDOM, 120, 84, 128, 96);
    prep_wt<<<dim3(18),   dim3(256), 0, stream>>>(cls_w3, w3t, NDOM, 84, 10, 96, 16);
    prep_wt<<<dim3(400),  dim3(256), 0, stream>>>(disc_w1, dw1t, 1, 784, 100, 800, 128);
    prep_wt<<<dim3(8),    dim3(256), 0, stream>>>(disc_w2, dw2t, 1, 100, 3, 128, 16);

    // conv1 stats -> BN1 params
    conv1_stats<<<dim3(BATCH), dim3(256), 0, stream>>>(x, conv1_w, conv1_b, ps1s, ps1q);
    bn_finalize<<<dim3(NDOM, 6), dim3(256), 0, stream>>>(ps1s, ps1q, dom, bn1_g, bn1_b,
                                                         sc1, sh1, 6, 784.0f);
    // conv1 apply (+pool, store h1p) + conv2 stats -> BN2 params
    conv1_apply_conv2_stats<<<dim3(BATCH), dim3(256), 0, stream>>>(
        x, dom, conv1_w, conv1_b, sc1, sh1, conv2_w, conv2_b, h1p, ps2s, ps2q);
    bn_finalize<<<dim3(NDOM, 16), dim3(256), 0, stream>>>(ps2s, ps2q, dom, bn2_g, bn2_b,
                                                          sc2, sh2, 16, 196.0f);
    // conv2 apply -> feats (f16, padded K=800)
    conv2_apply<<<dim3(BATCH), dim3(256), 0, stream>>>(h1p, dom, conv2_w, conv2_b,
                                                       sc2, sh2, feats);

    // Classifier: all 3 domains, then gather (matches reference einsum + take_along_axis)
    gemm_relu_f16<<<dim3(BATCH / 16, 8, NDOM), dim3(32), 0, stream>>>(
        feats, 0, 800, w1t, 128LL * 800, 800, cls_b1, 120, 120,
        a1, (long long)BATCH * 128, 128, 25);
    gemm_relu_f16<<<dim3(BATCH / 16, 6, NDOM), dim3(32), 0, stream>>>(
        a1, (long long)BATCH * 128, 128, w2t, 96LL * 128, 128, cls_b2, 84, 84,
        a2, (long long)BATCH * 96, 96, 4);
    gemm_logits_gather<<<dim3(BATCH / 16, 1, NDOM), dim3(32), 0, stream>>>(
        a2, (long long)BATCH * 96, 96, w3t, 16LL * 96, 96, cls_b3, 10, dom,
        out, 10, 3);

    // Discriminator
    gemm_relu_f16<<<dim3(BATCH / 16, 8, 1), dim3(32), 0, stream>>>(
        feats, 0, 800, dw1t, 0, 800, disc_b1, 0, 100, hd, 0, 128, 25);
    gemm_out_f32<<<dim3(BATCH / 16, 1, 1), dim3(32), 0, stream>>>(
        hd, 128, dw2t, 128, disc_b2, out + (size_t)BATCH * 10, 3, 4);
}
